// MusicVAE_72730976190578
// MI455X (gfx1250) — matrix operand order repaired and checked by reading, not compiled, not run on hardware
//
#include <hip/hip_runtime.h>
#include <hip/hip_bf16.h>
#include <math.h>

typedef _Float16 h16;
typedef __attribute__((ext_vector_type(8)))  _Float16 v8h;
typedef __attribute__((ext_vector_type(16))) _Float16 v16h;
typedef __attribute__((ext_vector_type(8)))  float    v8f;

#define BB 256
#define TT 64
#define DD 512
#define HEnc 1024
#define LL 512
#define NBAR 4
#define BARU 16

static __device__ __forceinline__ float sigmoidf_(float x) { return 1.f / (1.f + __expf(-x)); }

#define WMMA16(a, b, c) __builtin_amdgcn_wmma_f32_16x16x32_f16(false, (a), false, (b), (short)0, (c), false, false)

// ----------------------------------------------------------------------------
// CDNA5 async copy: global -> LDS, 16 bytes per lane, tracked by ASYNCcnt.
// VDST VGPR holds the LDS byte address (low 32 bits of a generic LDS pointer
// per the flat-aperture rule); VADDR holds the 64-bit
// global address; no SADDR ("off").
// ----------------------------------------------------------------------------
static __device__ __forceinline__ void async_cp16(const h16* g, const h16* s) {
  unsigned lds = (unsigned)(unsigned long long)(uintptr_t)s;
  asm volatile("global_load_async_to_lds_b128 %0, %1, off" :: "v"(lds), "v"(g) : "memory");
}
static __device__ __forceinline__ void wait_async_le4() {
  asm volatile("s_wait_asynccnt 0x4" ::: "memory");
}
static __device__ __forceinline__ void wait_async_0() {
  asm volatile("s_wait_asynccnt 0x0" ::: "memory");
}

// ----------------------------------------------------------------------------
// Generic convert / fill / copy kernels
// ----------------------------------------------------------------------------
__global__ void k_cvt_f32_f16(const float* __restrict__ in, h16* __restrict__ out, int n) {
  int i = blockIdx.x * blockDim.x + threadIdx.x;
  if (i < n) out[i] = (h16)in[i];
}

__global__ void k_fill_f32(float* __restrict__ p, float v, int n) {
  int i = blockIdx.x * blockDim.x + threadIdx.x;
  if (i < n) p[i] = v;
}

__global__ void k_fill_f16(h16* __restrict__ p, float v, int n) {
  int i = blockIdx.x * blockDim.x + threadIdx.x;
  if (i < n) p[i] = (h16)v;
}

// src (rows, cols) f32 row-major -> dst f16 with row stride + column offset
__global__ void k_cvt_strided(const float* __restrict__ src, h16* __restrict__ dst,
                              int rows, int cols, int dstStride, int dstOff) {
  int i = blockIdx.x * blockDim.x + threadIdx.x;
  if (i >= rows * cols) return;
  int r = i / cols, c = i % cols;
  dst[(size_t)r * dstStride + dstOff + c] = (h16)src[i];
}

// x (B,T,D) f32 -> xt (T,B,D) f16
__global__ void k_xpose(const float* __restrict__ x, h16* __restrict__ xt) {
  int i = blockIdx.x * blockDim.x + threadIdx.x;
  if (i >= BB * TT * DD) return;
  int d = i % DD;
  int t = (i / DD) % TT;
  int b = i / (DD * TT);
  xt[((size_t)t * BB + b) * DD + d] = (h16)x[i];
}

// ----------------------------------------------------------------------------
// WMMA GEMM:  C[M,N] (f32) = A[M,K](f16, row-major) @ W[N,K](f16, row-major)^T
// Block = 128 threads = 4 waves; block tile 64x64; wave tile 32x32 (2x2 frags).
// K-tiles of 32 are double-buffered in LDS and filled with
// global_load_async_to_lds_b128 (ASYNCcnt), so global latency is decoupled
// from the v_wmma issue stream. Async loads complete in order per wave:
// steady-state wait is asynccnt<=4 (one newer tile in flight).
// Fragment layouts used here (A and B share one contiguous K order):
//   A-frag lane l: row = (l&15), 16 consecutive K at 16*(l>>4)
//   B-frag lane l: col = (l&15), 16 consecutive K at 16*(l>>4)  (one 32B LDS read)
//   C-frag VGPR r: row = 8*(l>>4) + r, col = (l&15)
// ----------------------------------------------------------------------------
static __device__ __forceinline__ void store_acc(float* __restrict__ C, int N, v8f acc,
                                                 int row, int col, int accum) {
#pragma unroll
  for (int r = 0; r < 8; r++) {
    float* cp = C + (size_t)(row + r) * N + col;
    if (accum) *cp += acc[r]; else *cp = acc[r];
  }
}

__global__ __launch_bounds__(128) void k_gemm_wmma(const h16* __restrict__ A,
                                                   const h16* __restrict__ W,
                                                   float* __restrict__ C,
                                                   int M, int N, int K, int accum) {
  __shared__ __align__(32) h16 sA[2][64 * 32];
  __shared__ __align__(32) h16 sB[2][64 * 32];

  const int tid  = threadIdx.x;
  const int lane = tid & 31;
  const int wave = tid >> 5;
  const int m0blk = blockIdx.y * 64;
  const int n0blk = blockIdx.x * 64;
  const int wm = (wave >> 1) * 32;   // wave tile offsets inside the 64x64 block tile
  const int wn = (wave & 1) * 32;
  const int l15 = lane & 15;
  const int hi  = lane >> 4;

  // Staging: tile = 64 rows x 32 k x f16 = 4KB = 256 chunks of 16B.
  // 128 threads -> 2 chunks each for A and for B.
  const int c0 = tid, c1 = tid + 128;
  const int r0 = c0 >> 2, s0 = (c0 & 3) * 8;   // row, element offset within row
  const int r1 = c1 >> 2, s1 = (c1 & 3) * 8;
  const h16* gA0 = A + (size_t)(m0blk + r0) * K + s0;
  const h16* gA1 = A + (size_t)(m0blk + r1) * K + s1;
  const h16* gB0 = W + (size_t)(n0blk + r0) * K + s0;
  const h16* gB1 = W + (size_t)(n0blk + r1) * K + s1;

#define ISSUE_TILE(buf, kk)                                  \
  do {                                                       \
    async_cp16(gA0 + (kk), &sA[(buf)][r0 * 32 + s0]);        \
    async_cp16(gA1 + (kk), &sA[(buf)][r1 * 32 + s1]);        \
    async_cp16(gB0 + (kk), &sB[(buf)][r0 * 32 + s0]);        \
    async_cp16(gB1 + (kk), &sB[(buf)][r1 * 32 + s1]);        \
  } while (0)

  const int KT = K >> 5;
  ISSUE_TILE(0, 0);
  if (KT > 1) ISSUE_TILE(1, 32);

  v8f acc00 = {}, acc01 = {}, acc10 = {}, acc11 = {};

  for (int kt = 0; kt < KT; ++kt) {
    const int cur = kt & 1;
    // ensure tile kt has landed in LDS (in-order completion per wave)
    if (kt + 1 < KT) wait_async_le4(); else wait_async_0();
    __syncthreads();   // all waves' chunks of tile kt visible

    const h16* a0p = &sA[cur][(wm + l15) * 32 + hi * 16];
    const h16* a1p = a0p + 16 * 32;
    v8h a0lo = *(const v8h*)(a0p);
    v8h a0hi = *(const v8h*)(a0p + 8);
    v8h a1lo = *(const v8h*)(a1p);
    v8h a1hi = *(const v8h*)(a1p + 8);
    v16h a0 = __builtin_shufflevector(a0lo, a0hi, 0, 1, 2, 3, 4, 5, 6, 7,
                                      8, 9, 10, 11, 12, 13, 14, 15);
    v16h a1 = __builtin_shufflevector(a1lo, a1hi, 0, 1, 2, 3, 4, 5, 6, 7,
                                      8, 9, 10, 11, 12, 13, 14, 15);
    const h16* b0p = &sB[cur][(wn + l15) * 32 + hi * 16];
    v16h b0 = *(const v16h*)(b0p);
    v16h b1 = *(const v16h*)(b0p + 16 * 32);

    acc00 = WMMA16(a0, b0, acc00);
    acc01 = WMMA16(a0, b1, acc01);
    acc10 = WMMA16(a1, b0, acc10);
    acc11 = WMMA16(a1, b1, acc11);

    __syncthreads();   // all waves finished reading buffer `cur`
    if (kt + 2 < KT) ISSUE_TILE(cur, (kt + 2) * 32);   // safe to overwrite
  }
#undef ISSUE_TILE

  const int row0 = m0blk + wm + hi * 8;
  const int row1 = m0blk + wm + 16 + hi * 8;
  const int col0 = n0blk + wn + l15;
  const int col1 = n0blk + wn + 16 + l15;
  store_acc(C, N, acc00, row0, col0, accum);
  store_acc(C, N, acc01, row0, col1, accum);
  store_acc(C, N, acc10, row1, col0, accum);
  store_acc(C, N, acc11, row1, col1, accum);
}

// ----------------------------------------------------------------------------
// LSTM pointwise: gates (B,4H) f32 (sum of the two GEMMs), biases bih+bhh,
// update c, h (f32) and emit h in f16 for the next-step WMMA.
// torch gate order: i, f, g, o.
// ----------------------------------------------------------------------------
__global__ __launch_bounds__(256) void k_lstm_pw(const float* __restrict__ gates,
                                                 const float* __restrict__ bih,
                                                 const float* __restrict__ bhh,
                                                 float* __restrict__ c,
                                                 float* __restrict__ h,
                                                 h16* __restrict__ hf16, int H) {
  int idx = blockIdx.x * blockDim.x + threadIdx.x;
  if (idx >= BB * H) return;
  int b = idx / H, j = idx % H;
  const float* g = gates + (size_t)b * 4 * H;
  float gi = g[j]         + bih[j]         + bhh[j];
  float gf = g[H + j]     + bih[H + j]     + bhh[H + j];
  float gg = g[2 * H + j] + bih[2 * H + j] + bhh[2 * H + j];
  float go = g[3 * H + j] + bih[3 * H + j] + bhh[3 * H + j];
  float cc = sigmoidf_(gf) * c[idx] + sigmoidf_(gi) * tanhf(gg);
  float hh = sigmoidf_(go) * tanhf(cc);
  c[idx] = cc;
  h[idx] = hh;
  hf16[idx] = (h16)hh;
}

// ----------------------------------------------------------------------------
// Latent head: mu = normalize(mu_raw + bmu), std = softplus, z = mu + eps*std.
// One block (256 thr) per batch row, L = 512.
// ----------------------------------------------------------------------------
__global__ __launch_bounds__(256) void k_latent(const float* __restrict__ mu_raw,
                                                const float* __restrict__ std_raw,
                                                const float* __restrict__ bmu,
                                                const float* __restrict__ bstd,
                                                const float* __restrict__ eps,
                                                float* __restrict__ out_mu,
                                                float* __restrict__ out_std,
                                                float* __restrict__ z,
                                                h16* __restrict__ zf16) {
  __shared__ float red[256];
  int b = blockIdx.x, tid = threadIdx.x;
  size_t base = (size_t)b * LL;
  float x0 = mu_raw[base + tid] + bmu[tid];
  float x1 = mu_raw[base + 256 + tid] + bmu[256 + tid];
  red[tid] = x0 + x1;
  __syncthreads();
  for (int s = 128; s > 0; s >>= 1) {
    if (tid < s) red[tid] += red[tid + s];
    __syncthreads();
  }
  float mean = red[0] * (1.f / LL);
  __syncthreads();
  float d0 = x0 - mean, d1 = x1 - mean;
  red[tid] = d0 * d0 + d1 * d1;
  __syncthreads();
  for (int s = 128; s > 0; s >>= 1) {
    if (tid < s) red[tid] += red[tid + s];
    __syncthreads();
  }
  float var = red[0] * (1.f / LL);
  float inv = rsqrtf(var + 1e-5f);
  float m0 = d0 * inv, m1 = d1 * inv;
  out_mu[base + tid] = m0;
  out_mu[base + 256 + tid] = m1;
  float s0 = std_raw[base + tid] + bstd[tid];
  float s1 = std_raw[base + 256 + tid] + bstd[256 + tid];
  float sp0 = (s0 > 20.f) ? s0 : log1pf(__expf(s0));
  float sp1 = (s1 > 20.f) ? s1 : log1pf(__expf(s1));
  out_std[base + tid] = sp0;
  out_std[base + 256 + tid] = sp1;
  float z0 = m0 + eps[base + tid] * sp0;
  float z1 = m1 + eps[base + 256 + tid] * sp1;
  z[base + tid] = z0;
  z[base + 256 + tid] = z1;
  zf16[base + tid] = (h16)z0;
  zf16[base + 256 + tid] = (h16)z1;
}

// Conductor init: h0=c0=h1=c1=z, plus f16 copies of the recurrent states.
__global__ void k_cond_init(const float* __restrict__ z, float* h0, float* c0,
                            float* h1, float* c1, h16* h0f, h16* h1f) {
  int i = blockIdx.x * blockDim.x + threadIdx.x;
  if (i >= BB * LL) return;
  float v = z[i];
  h0[i] = v; c0[i] = v; h1[i] = v; c1[i] = v;
  h0f[i] = (h16)v; h1f[i] = (h16)v;
}

// feat = featc + bcl  (rows = B*NBAR laid out as (b,bar))
__global__ void k_feat_bias(const float* __restrict__ featc, const float* __restrict__ bcl,
                            float* __restrict__ featb) {
  int i = blockIdx.x * blockDim.x + threadIdx.x;
  if (i >= BB * NBAR * LL) return;
  featb[i] = featc[i] + bcl[i % LL];
}

// Decoder bar reset: hd*/cd* = feat[b,bar]; also fill xin[:, D:D+L] with f16 z-embed.
__global__ void k_dec_reset(const float* __restrict__ featb, float* hd1, float* cd1,
                            float* hd2, float* cd2, h16* h1f, h16* h2f,
                            h16* __restrict__ xin, int bar) {
  int i = blockIdx.x * blockDim.x + threadIdx.x;
  if (i >= BB * LL) return;
  int b = i / LL, j = i % LL;
  float v = featb[((size_t)b * NBAR + bar) * LL + j];
  hd1[i] = v; cd1[i] = v; hd2[i] = v; cd2[i] = v;
  h16 hv = (h16)v;
  h1f[i] = hv; h2f[i] = hv;
  xin[(size_t)b * (DD + LL) + DD + j] = hv;
}

// xin[:, 0:D] = x_prev (Xf16[t-1]) or zeros at t==0
__global__ void k_xprev(h16* __restrict__ xin, const h16* __restrict__ Xf16, int t) {
  int i = blockIdx.x * blockDim.x + threadIdx.x;
  if (i >= BB * DD) return;
  int b = i / DD, j = i % DD;
  h16 v = (t == 0) ? (h16)0.f : Xf16[((size_t)(t - 1) * BB + b) * DD + j];
  xin[(size_t)b * (DD + LL) + j] = v;
}

// Softmax + argmax over D=512 logits for one timestep; writes prob slice + label.
__global__ __launch_bounds__(256) void k_softmax_label(const float* __restrict__ logits,
                                                       const float* __restrict__ blog,
                                                       float* __restrict__ prob,
                                                       float* __restrict__ label, int t) {
  __shared__ float red[256];
  __shared__ int ridx[256];
  int b = blockIdx.x, tid = threadIdx.x;
  size_t base = (size_t)b * DD;
  float x0 = logits[base + tid] + blog[tid];
  float x1 = logits[base + 256 + tid] + blog[256 + tid];
  red[tid] = fmaxf(x0, x1);
  __syncthreads();
  for (int s = 128; s > 0; s >>= 1) {
    if (tid < s) red[tid] = fmaxf(red[tid], red[tid + s]);
    __syncthreads();
  }
  float mx = red[0];
  __syncthreads();
  float e0 = __expf(x0 - mx), e1 = __expf(x1 - mx);
  red[tid] = e0 + e1;
  __syncthreads();
  for (int s = 128; s > 0; s >>= 1) {
    if (tid < s) red[tid] += red[tid + s];
    __syncthreads();
  }
  float inv = 1.f / red[0];
  __syncthreads();
  size_t pbase = ((size_t)b * TT + t) * DD;
  prob[pbase + tid] = e0 * inv;
  prob[pbase + 256 + tid] = e1 * inv;
  // argmax (first occurrence wins on ties; tid < tid+256 so >= keeps earlier)
  float v; int ix;
  if (x0 >= x1) { v = x0; ix = tid; } else { v = x1; ix = tid + 256; }
  red[tid] = v; ridx[tid] = ix;
  __syncthreads();
  for (int s = 128; s > 0; s >>= 1) {
    if (tid < s) {
      if (red[tid + s] > red[tid] ||
          (red[tid + s] == red[tid] && ridx[tid + s] < ridx[tid])) {
        red[tid] = red[tid + s];
        ridx[tid] = ridx[tid + s];
      }
    }
    __syncthreads();
  }
  if (tid == 0) label[(size_t)b * TT + t] = (float)ridx[0];
}

// ----------------------------------------------------------------------------
// Host orchestration
// ----------------------------------------------------------------------------
extern "C" void kernel_launch(void* const* d_in, const int* in_sizes, int n_in,
                              void* d_out, int out_size, void* d_ws, size_t ws_size,
                              hipStream_t stream) {
  const float* x       = (const float*)d_in[0];
  const float* eps     = (const float*)d_in[1];
  const float* We_ih_f = (const float*)d_in[2];
  const float* We_hh_f = (const float*)d_in[3];
  const float* be_ih_f = (const float*)d_in[4];
  const float* be_hh_f = (const float*)d_in[5];
  const float* We_ih_b = (const float*)d_in[6];
  const float* We_hh_b = (const float*)d_in[7];
  const float* be_ih_b = (const float*)d_in[8];
  const float* be_hh_b = (const float*)d_in[9];
  const float* Wmu     = (const float*)d_in[10];
  const float* bmu     = (const float*)d_in[11];
  const float* Wstd    = (const float*)d_in[12];
  const float* bstd    = (const float*)d_in[13];
  const float* Wc_ih0  = (const float*)d_in[14];
  const float* Wc_hh0  = (const float*)d_in[15];
  const float* bc_ih0  = (const float*)d_in[16];
  const float* bc_hh0  = (const float*)d_in[17];
  const float* Wc_ih1  = (const float*)d_in[18];
  const float* Wc_hh1  = (const float*)d_in[19];
  const float* bc_ih1  = (const float*)d_in[20];
  const float* bc_hh1  = (const float*)d_in[21];
  const float* Wcl     = (const float*)d_in[22];
  const float* bcl     = (const float*)d_in[23];
  const float* Wd_ih0  = (const float*)d_in[24];
  const float* Wd_hh0  = (const float*)d_in[25];
  const float* bd_ih0  = (const float*)d_in[26];
  const float* bd_hh0  = (const float*)d_in[27];
  const float* Wd_ih1  = (const float*)d_in[28];
  const float* Wd_hh1  = (const float*)d_in[29];
  const float* bd_ih1  = (const float*)d_in[30];
  const float* bd_hh1  = (const float*)d_in[31];
  const float* Wlog    = (const float*)d_in[32];
  const float* blog    = (const float*)d_in[33];

  float* out       = (float*)d_out;
  float* out_prob  = out;
  float* out_mu    = out + (size_t)BB * TT * DD;
  float* out_std   = out_mu + (size_t)BB * LL;
  float* out_label = out_std + (size_t)BB * LL;

  // bump allocator over d_ws
  size_t off = 0;
  auto alloc = [&](size_t bytes) -> void* {
    void* p = (char*)d_ws + off;
    off += (bytes + 255) & ~(size_t)255;
    return p;
  };

  // f16 weights
  h16* w_eihf = (h16*)alloc((size_t)4 * HEnc * DD * 2);
  h16* w_ehhf = (h16*)alloc((size_t)4 * HEnc * HEnc * 2);
  h16* w_eihb = (h16*)alloc((size_t)4 * HEnc * DD * 2);
  h16* w_ehhb = (h16*)alloc((size_t)4 * HEnc * HEnc * 2);
  h16* w_mu   = (h16*)alloc((size_t)LL * 2 * HEnc * 2);
  h16* w_std  = (h16*)alloc((size_t)LL * 2 * HEnc * 2);
  h16* w_cih0 = (h16*)alloc((size_t)4 * LL * LL * 2);
  h16* w_chh0 = (h16*)alloc((size_t)4 * LL * LL * 2);
  h16* w_cih1 = (h16*)alloc((size_t)4 * LL * LL * 2);
  h16* w_chh1 = (h16*)alloc((size_t)4 * LL * LL * 2);
  h16* w_cl   = (h16*)alloc((size_t)LL * LL * 2);
  h16* w_dih0 = (h16*)alloc((size_t)4 * LL * (LL + DD) * 2);
  h16* w_dhh0 = (h16*)alloc((size_t)4 * LL * LL * 2);
  h16* w_dih1 = (h16*)alloc((size_t)4 * LL * LL * 2);
  h16* w_dhh1 = (h16*)alloc((size_t)4 * LL * LL * 2);
  h16* w_log  = (h16*)alloc((size_t)DD * LL * 2);

  // activations
  h16*   Xf16    = (h16*)alloc((size_t)TT * BB * DD * 2);
  float* gates   = (float*)alloc((size_t)BB * 4 * HEnc * 4);
  float* h_enc   = (float*)alloc((size_t)BB * HEnc * 4);
  float* c_enc   = (float*)alloc((size_t)BB * HEnc * 4);
  h16*   hf16e   = (h16*)alloc((size_t)BB * HEnc * 2);
  h16*   hencf16 = (h16*)alloc((size_t)BB * 2 * HEnc * 2);
  float* mu_raw  = (float*)alloc((size_t)BB * LL * 4);
  float* std_raw = (float*)alloc((size_t)BB * LL * 4);
  float* zf32    = (float*)alloc((size_t)BB * LL * 4);
  h16*   zf16    = (h16*)alloc((size_t)BB * LL * 2);
  float* ch0     = (float*)alloc((size_t)BB * LL * 4);
  float* cc0     = (float*)alloc((size_t)BB * LL * 4);
  float* ch1     = (float*)alloc((size_t)BB * LL * 4);
  float* cc1     = (float*)alloc((size_t)BB * LL * 4);
  h16*   h0f16   = (h16*)alloc((size_t)BB * LL * 2);
  h16*   h1f16   = (h16*)alloc((size_t)BB * LL * 2);
  h16*   featsA  = (h16*)alloc((size_t)BB * NBAR * LL * 2);
  float* featc   = (float*)alloc((size_t)BB * NBAR * LL * 4);
  float* featb   = (float*)alloc((size_t)BB * NBAR * LL * 4);
  float* hd1     = (float*)alloc((size_t)BB * LL * 4);
  float* cd1     = (float*)alloc((size_t)BB * LL * 4);
  float* hd2     = (float*)alloc((size_t)BB * LL * 4);
  float* cd2     = (float*)alloc((size_t)BB * LL * 4);
  h16*   hd1f16  = (h16*)alloc((size_t)BB * LL * 2);
  h16*   hd2f16  = (h16*)alloc((size_t)BB * LL * 2);
  h16*   xin     = (h16*)alloc((size_t)BB * (DD + LL) * 2);
  float* logits  = (float*)alloc((size_t)BB * DD * 4);

  auto cvt = [&](const float* src, h16* dst, int n) {
    k_cvt_f32_f16<<<(n + 255) / 256, 256, 0, stream>>>(src, dst, n);
  };
  auto gemm = [&](const h16* A, const h16* W, float* C, int M, int N, int K, int accum) {
    dim3 g(N / 64, M / 64);
    k_gemm_wmma<<<g, 128, 0, stream>>>(A, W, C, M, N, K, accum);
  };
  auto fillf = [&](float* p, int n) {
    k_fill_f32<<<(n + 255) / 256, 256, 0, stream>>>(p, 0.f, n);
  };
  auto fillh = [&](h16* p, int n) {
    k_fill_f16<<<(n + 255) / 256, 256, 0, stream>>>(p, 0.f, n);
  };

  // ---- weight conversions (f32 -> f16) ----
  cvt(We_ih_f, w_eihf, 4 * HEnc * DD);
  cvt(We_hh_f, w_ehhf, 4 * HEnc * HEnc);
  cvt(We_ih_b, w_eihb, 4 * HEnc * DD);
  cvt(We_hh_b, w_ehhb, 4 * HEnc * HEnc);
  cvt(Wmu,  w_mu,  LL * 2 * HEnc);
  cvt(Wstd, w_std, LL * 2 * HEnc);
  cvt(Wc_ih0, w_cih0, 4 * LL * LL);
  cvt(Wc_hh0, w_chh0, 4 * LL * LL);
  cvt(Wc_ih1, w_cih1, 4 * LL * LL);
  cvt(Wc_hh1, w_chh1, 4 * LL * LL);
  cvt(Wcl, w_cl, LL * LL);
  cvt(Wd_ih0, w_dih0, 4 * LL * (LL + DD));
  cvt(Wd_hh0, w_dhh0, 4 * LL * LL);
  cvt(Wd_ih1, w_dih1, 4 * LL * LL);
  cvt(Wd_hh1, w_dhh1, 4 * LL * LL);
  cvt(Wlog, w_log, DD * LL);

  // ---- x -> (T,B,D) f16 ----
  k_xpose<<<(BB * TT * DD + 255) / 256, 256, 0, stream>>>(x, Xf16);

  // ---- Encoder forward ----
  fillf(h_enc, BB * HEnc); fillf(c_enc, BB * HEnc); fillh(hf16e, BB * HEnc);
  for (int t = 0; t < TT; t++) {
    gemm(Xf16 + (size_t)t * BB * DD, w_eihf, gates, BB, 4 * HEnc, DD, 0);
    gemm(hf16e, w_ehhf, gates, BB, 4 * HEnc, HEnc, 1);
    k_lstm_pw<<<(BB * HEnc) / 256, 256, 0, stream>>>(gates, be_ih_f, be_hh_f,
                                                     c_enc, h_enc, hf16e, HEnc);
  }
  k_cvt_strided<<<(BB * HEnc) / 256, 256, 0, stream>>>(h_enc, hencf16, BB, HEnc,
                                                       2 * HEnc, 0);

  // ---- Encoder backward ----
  fillf(h_enc, BB * HEnc); fillf(c_enc, BB * HEnc); fillh(hf16e, BB * HEnc);
  for (int t = 0; t < TT; t++) {
    gemm(Xf16 + (size_t)(TT - 1 - t) * BB * DD, w_eihb, gates, BB, 4 * HEnc, DD, 0);
    gemm(hf16e, w_ehhb, gates, BB, 4 * HEnc, HEnc, 1);
    k_lstm_pw<<<(BB * HEnc) / 256, 256, 0, stream>>>(gates, be_ih_b, be_hh_b,
                                                     c_enc, h_enc, hf16e, HEnc);
  }
  k_cvt_strided<<<(BB * HEnc) / 256, 256, 0, stream>>>(h_enc, hencf16, BB, HEnc,
                                                       2 * HEnc, HEnc);

  // ---- Latent ----
  gemm(hencf16, w_mu,  mu_raw,  BB, LL, 2 * HEnc, 0);
  gemm(hencf16, w_std, std_raw, BB, LL, 2 * HEnc, 0);
  k_latent<<<BB, 256, 0, stream>>>(mu_raw, std_raw, bmu, bstd, eps,
                                   out_mu, out_std, zf32, zf16);

  // ---- Conductor (2-layer LSTM over NBAR steps, h0=c0=z, input z) ----
  k_cond_init<<<(BB * LL) / 256, 256, 0, stream>>>(zf32, ch0, cc0, ch1, cc1,
                                                   h0f16, h1f16);
  for (int bar = 0; bar < NBAR; bar++) {
    gemm(zf16,  w_cih0, gates, BB, 4 * LL, LL, 0);
    gemm(h0f16, w_chh0, gates, BB, 4 * LL, LL, 1);
    k_lstm_pw<<<(BB * LL) / 256, 256, 0, stream>>>(gates, bc_ih0, bc_hh0,
                                                   cc0, ch0, h0f16, LL);
    gemm(h0f16, w_cih1, gates, BB, 4 * LL, LL, 0);
    gemm(h1f16, w_chh1, gates, BB, 4 * LL, LL, 1);
    k_lstm_pw<<<(BB * LL) / 256, 256, 0, stream>>>(gates, bc_ih1, bc_hh1,
                                                   cc1, ch1, h1f16, LL);
    k_cvt_strided<<<(BB * LL) / 256, 256, 0, stream>>>(ch1, featsA, BB, LL,
                                                       NBAR * LL, bar * LL);
  }

  // ---- feat = stack(h1) @ Wcl^T + bcl  (rows = (b,bar)) ----
  gemm(featsA, w_cl, featc, BB * NBAR, LL, LL, 0);
  k_feat_bias<<<(BB * NBAR * LL) / 256, 256, 0, stream>>>(featc, bcl, featb);

  // ---- Decoder: 2-layer LSTM, teacher-forced, reset each bar ----
  for (int bar = 0; bar < NBAR; bar++) {
    k_dec_reset<<<(BB * LL) / 256, 256, 0, stream>>>(featb, hd1, cd1, hd2, cd2,
                                                     hd1f16, hd2f16, xin, bar);
    for (int u = 0; u < BARU; u++) {
      int t = bar * BARU + u;
      k_xprev<<<(BB * DD) / 256, 256, 0, stream>>>(xin, Xf16, t);
      gemm(xin,    w_dih0, gates, BB, 4 * LL, DD + LL, 0);
      gemm(hd1f16, w_dhh0, gates, BB, 4 * LL, LL, 1);
      k_lstm_pw<<<(BB * LL) / 256, 256, 0, stream>>>(gates, bd_ih0, bd_hh0,
                                                     cd1, hd1, hd1f16, LL);
      gemm(hd1f16, w_dih1, gates, BB, 4 * LL, LL, 0);
      gemm(hd2f16, w_dhh1, gates, BB, 4 * LL, LL, 1);
      k_lstm_pw<<<(BB * LL) / 256, 256, 0, stream>>>(gates, bd_ih1, bd_hh1,
                                                     cd2, hd2, hd2f16, LL);
      gemm(hd2f16, w_log, logits, BB, DD, LL, 0);
      k_softmax_label<<<BB, 256, 0, stream>>>(logits, blog, out_prob, out_label, t);
    }
  }
  (void)in_sizes; (void)n_in; (void)out_size; (void)ws_size;
}
